// FullGRUODECell_Autonomous_fine_90211493085353
// MI455X (gfx1250) — matrix operand order repaired and checked by reading, not compiled, not run on hardware
//
#include <hip/hip_runtime.h>
#include <hip/hip_bf16.h>

typedef __attribute__((ext_vector_type(16))) _Float16 v16h;
typedef __attribute__((ext_vector_type(8)))  _Float16 v8h;
typedef __attribute__((ext_vector_type(4)))  _Float16 v4h;
typedef __attribute__((ext_vector_type(8)))  float    v8f;

#define HD    256
#define BATCH 1024
#define KFINE 65536  // 256*256

union FragH { v16h v; v8h h[2]; };

// ---- WMMA fragment loaders (wave32, 16x16x32 f16; contiguous k order used for both A and B) ----
// A (16x32, MxK) from row-major X[M,K]: lane L holds M = m0 + (L&15);
// halves[0..15] = K = k0 + (L>>4)*16 + {0..15}
__device__ __forceinline__ v16h load_afrag(const _Float16* __restrict__ X, int ldk,
                                           int m0, int k0, int lane) {
    const int m = m0 + (lane & 15);
    const int g = lane >> 4;
    const _Float16* p = X + (size_t)m * ldk + k0 + g * 16;
    FragH u;
    u.h[0] = *(const v8h*)(p);
    u.h[1] = *(const v8h*)(p + 8);
    return u.v;
}

// B (32x16, KxN) for X @ W^T with row-major W[N,K]: lane L holds N = n0 + (L&15);
// halves[0..15] = K = k0 + (L>>4)*16 + {0..15}  -> 16 contiguous halves of row N.
__device__ __forceinline__ v16h load_bfrag(const _Float16* __restrict__ W, int ldk,
                                           int n0, int k0, int lane) {
    const int n = n0 + (lane & 15);
    const int g = lane >> 4;
    const _Float16* p = W + (size_t)n * ldk + k0 + g * 16;
    FragH u;
    u.h[0] = *(const v8h*)(p);
    u.h[1] = *(const v8h*)(p + 8);
    return u.v;
}

__device__ __forceinline__ v8f wmma_f16(v16h a, v16h b, v8f c) {
    return __builtin_amdgcn_wmma_f32_16x16x32_f16(false, a, false, b, (short)0, c,
                                                  false, false);
}

// ---- f32 -> f16 conversion ----
__global__ __launch_bounds__(256) void k_cvt(const float* __restrict__ src,
                                             _Float16* __restrict__ dst, int n) {
    int i = (blockIdx.x * blockDim.x + threadIdx.x) * 4;
    if (i + 3 < n) {
        float4 v = *(const float4*)(src + i);
        v4h o = { (_Float16)v.x, (_Float16)v.y, (_Float16)v.z, (_Float16)v.w };
        *(v4h*)(dst + i) = o;
    }
}

// ---- r-gate: rh = sigmoid(h @ W_hr^T) * h  (stored f16) ----
__global__ __launch_bounds__(256) void k_gate_r(const float* __restrict__ hf32,
                                                const _Float16* __restrict__ h16,
                                                const _Float16* __restrict__ whr16,
                                                _Float16* __restrict__ rh16) {
    const int lane = threadIdx.x & 31;
    const int wid  = (blockIdx.x * blockDim.x + threadIdx.x) >> 5;
    const int b0   = (wid >> 4) << 4;
    const int n0   = (wid & 15) << 4;

    v8f acc = {};
#pragma unroll
    for (int k0 = 0; k0 < HD; k0 += 32) {
        v16h a = load_afrag(h16, HD, b0, k0, lane);
        v16h b = load_bfrag(whr16, HD, n0, k0, lane);
        acc = wmma_f16(a, b, acc);
    }
    const int n  = n0 + (lane & 15);
    const int mb = b0 + ((lane >> 4) << 3);
#pragma unroll
    for (int r = 0; r < 8; ++r) {
        const int m = mb + r;
        float rg = 1.0f / (1.0f + __expf(-acc[r]));
        float hv = hf32[(size_t)m * HD + n];
        rh16[(size_t)m * HD + n] = (_Float16)(rg * hv);
    }
}

// ---- main fused kernel ----
// Block = 8 waves sharing one i-tile; wave w owns b-tile (bg*8+w).
// B tile for each j (16 V-rows x 256 halves = 8KB) is async-staged into LDS in
// fragment order and shared by all 8 waves (8x global-traffic reduction).
__global__ __launch_bounds__(256) void k_main(const float* __restrict__ hf32,
                                              const _Float16* __restrict__ h16,
                                              const _Float16* __restrict__ whz16,
                                              const _Float16* __restrict__ whh16,
                                              const _Float16* __restrict__ rh16,
                                              const _Float16* __restrict__ wf16,
                                              float* __restrict__ dh) {
    __shared__ _Float16 sh_b[2 * 16 * HD];   // double-buffered B tile, 2 x 8KB

    const int tid  = threadIdx.x;
    const int lane = tid & 31;
    const int w    = tid >> 5;               // wave in block 0..7
    const int i0   = (blockIdx.x & 15) << 4; // shared i-tile
    const int b0   = ((blockIdx.x >> 4) * 8 + w) << 4;  // per-wave b-tile

    // Async B loader: thread t stages one 32B chunk (two b128) per j.
    // Chunk = halves [mm*16, mm*16+16) of V row (i0+ri)  ==  fragment of lane
    // fl = ((mm&1)<<4)|ri for step kb = mm>>1.  Scatter to fragment-ordered LDS.
    const int ri  = tid >> 4;
    const int mm  = tid & 15;
    const int kbl = mm >> 1;
    const int fl  = ((mm & 1) << 4) | ri;
    const uint32_t dst0 = (uint32_t)(uintptr_t)(sh_b + kbl * 512 + fl * 16);
    const _Float16* gsrc0 = wf16 + (size_t)(i0 + ri) * KFINE + mm * 16;

    // prologue: stage j = 0 into buffer 0
    asm volatile("global_load_async_to_lds_b128 %0, %1, off"
                 :: "v"(dst0), "v"(gsrc0) : "memory");
    asm volatile("global_load_async_to_lds_b128 %0, %1, off offset:16"
                 :: "v"(dst0), "v"(gsrc0) : "memory");

    // Hoisted A fragments of this wave's h b-tile (z-GEMM + all 256 j-steps).
    v16h afrag[8];
#pragma unroll
    for (int kb = 0; kb < 8; ++kb)
        afrag[kb] = load_afrag(h16, HD, b0, kb * 32, lane);

    v8f accZ = {};
    v8f accU = {};
#pragma unroll
    for (int kb = 0; kb < 8; ++kb) {
        v16h bz = load_bfrag(whz16, HD, i0, kb * 32, lane);
        accZ = wmma_f16(afrag[kb], bz, accZ);
        v16h au = load_afrag(rh16, HD, b0, kb * 32, lane);
        v16h bu = load_bfrag(whh16, HD, i0, kb * 32, lane);
        accU = wmma_f16(au, bu, accU);
    }

    // Hyper GEMM: hz_small[b,i] = sum_j h[b,j] * (sum_k h16[b,k]*V[i, j*256+k])
    const int mboff = (lane >> 4) << 3;
    const int hrow  = b0 + mboff;            // first of this lane's 8 C rows

    for (int j = 0; j < HD; ++j) {
        // own chunks of buffer (j&1) complete ...
        asm volatile("s_wait_asynccnt 0x0" ::: "memory");
        // ... and everyone else's too (also retires reads of the other buffer)
        __syncthreads();
        if (j + 1 < HD) {
            const _Float16* gp = gsrc0 + (size_t)(j + 1) * HD;
            const uint32_t d = dst0 + (((j + 1) & 1) ? 8192u : 0u);
            asm volatile("global_load_async_to_lds_b128 %0, %1, off"
                         :: "v"(d), "v"(gp) : "memory");
            asm volatile("global_load_async_to_lds_b128 %0, %1, off offset:16"
                         :: "v"(d), "v"(gp) : "memory");
        }
        const _Float16* bbuf = sh_b + ((j & 1) ? 4096 : 0);
        v8f t = {};
#pragma unroll
        for (int kb = 0; kb < 8; ++kb) {
            const _Float16* bp = bbuf + kb * 512 + lane * 16;  // conflict-free
            FragH ub;
            ub.h[0] = *(const v8h*)(bp);
            ub.h[1] = *(const v8h*)(bp + 8);
            t = wmma_f16(afrag[kb], ub.v, t);
        }
#pragma unroll
        for (int r = 0; r < 8; ++r) {       // f32 scale by h[b_row, j] (L0-resident)
            float hj = hf32[(size_t)(hrow + r) * HD + j];
            accZ[r] = fmaf(hj, t[r], accZ[r]);
        }
    }

    // Epilogue: z = sigmoid(accZ); u = tanh(accU); dh = (1-z)*(u-h)
    const int n = i0 + (lane & 15);
#pragma unroll
    for (int r = 0; r < 8; ++r) {
        const int m = hrow + r;
        float z  = 1.0f / (1.0f + __expf(-accZ[r]));
        float u  = tanhf(accU[r]);
        float hv = hf32[(size_t)m * HD + n];
        dh[(size_t)m * HD + n] = (1.0f - z) * (u - hv);
    }
}

extern "C" void kernel_launch(void* const* d_in, const int* in_sizes, int n_in,
                              void* d_out, int out_size, void* d_ws, size_t ws_size,
                              hipStream_t stream) {
    // inputs: t(scalar, unused), h, W_hr, W_hz, W_hh, W_fine  (all f32)
    const float* h   = (const float*)d_in[1];
    const float* Whr = (const float*)d_in[2];
    const float* Whz = (const float*)d_in[3];
    const float* Whh = (const float*)d_in[4];
    const float* Wf  = (const float*)d_in[5];
    float* out = (float*)d_out;

    char* ws = (char*)d_ws;
    _Float16* wf16  = (_Float16*)(ws);                       // 33,554,432 B
    _Float16* h16   = (_Float16*)(ws + 33554432);
    _Float16* whr16 = (_Float16*)(ws + 34078720);
    _Float16* whz16 = (_Float16*)(ws + 34209792);
    _Float16* whh16 = (_Float16*)(ws + 34340864);
    _Float16* rh16  = (_Float16*)(ws + 34471936);

    auto cvt = [&](const float* s, _Float16* d, int n) {
        int blocks = (n / 4 + 255) / 256;
        k_cvt<<<blocks, 256, 0, stream>>>(s, d, n);
    };
    cvt(Wf,  wf16,  KFINE * HD);
    cvt(h,   h16,   BATCH * HD);
    cvt(Whr, whr16, HD * HD);
    cvt(Whz, whz16, HD * HD);
    cvt(Whh, whh16, HD * HD);

    k_gate_r<<<128, 256, 0, stream>>>(h, h16, whr16, rh16);
    // 128 blocks: blockIdx = bg*16 + ig ... (bg = blockIdx>>4, ig = blockIdx&15)
    k_main  <<<128, 256, 0, stream>>>(h, h16, whz16, whh16, rh16, wf16, out);
}
